// PharmHGT_6975026889045
// MI455X (gfx1250) — matrix operand order repaired and checked by reading, not compiled, not run on hardware
//
#include <hip/hip_runtime.h>
#include <cstddef>

// ---------------------------------------------------------------------------
// Types for CDNA5 WMMA (gfx1250, wave32)
// ---------------------------------------------------------------------------
typedef __bf16 bf16_t;
typedef __attribute__((ext_vector_type(16))) __bf16 v16bf;
typedef __attribute__((ext_vector_type(8)))  __bf16 v8bf;
typedef __attribute__((ext_vector_type(8)))  float  v8f;

#define HID   300
#define OPAD  320           // padded K for H=300 (mult of 32)
#define NB    256           // batch (graphs)
#define NA_G  128
#define NP_G  16
#define NAT   (NB * NA_G)   // 32768 atom nodes
#define NPH_N (NB * NP_G)   // 4096 pharm nodes
#define EA    (NAT * 4)     // 131072 bond edges
#define EP    (NPH_N * 2)   // 8192 reac edges

__device__ __forceinline__ bf16_t f2bf(float f) {
  unsigned int u = __builtin_bit_cast(unsigned int, f);
  unsigned int r = (u + 0x7fffu + ((u >> 16) & 1u)) >> 16;
  return __builtin_bit_cast(bf16_t, (unsigned short)r);
}
__device__ __forceinline__ float dsigmoid(float x) { return 1.0f / (1.0f + expf(-x)); }

__device__ __forceinline__ v16bf cat8(v8bf a, v8bf b) {
  return __builtin_shufflevector(a, b, 0, 1, 2, 3, 4, 5, 6, 7, 8, 9, 10, 11, 12, 13, 14, 15);
}

// ---------------------------------------------------------------------------
// f32 [R,K] -> zero-padded bf16 [Rpad?,Kpad] (rows >= R and cols >= K get 0)
// ---------------------------------------------------------------------------
__global__ void cvt_pad_kernel(const float* __restrict__ in, bf16_t* __restrict__ out,
                               int R, int K, int Kpad, int total) {
  int i = blockIdx.x * blockDim.x + threadIdx.x;
  if (i >= total) return;
  int r = i / Kpad, k = i % Kpad;
  float v = (r < R && k < K) ? in[(size_t)r * K + k] : 0.f;
  out[i] = f2bf(v);
}

__global__ void pad_zero_bf(bf16_t* __restrict__ dst, int M, int ld, int from) {
  int i = blockIdx.x * blockDim.x + threadIdx.x;
  int w = ld - from;
  if (i >= M * w) return;
  int r = i / w, c = i % w;
  dst[(size_t)r * ld + from + c] = f2bf(0.f);
}

// ---------------------------------------------------------------------------
// GEMM: C[M,N] = act( A[M,Kpad](bf16) * W[Npad,Kpad](bf16)^T + bias + res )
// M multiple of 16. One wave computes two adjacent 16x16 C tiles sharing the
// A fragment. Wave/tile indices forced scalar (readfirstlane) so tile validity
// is a scalar branch hoisted out of the K-loop -> branch-free WMMA inner loop.
// ---------------------------------------------------------------------------
__global__ __launch_bounds__(128) void gemm_kernel(
    const bf16_t* __restrict__ A, const bf16_t* __restrict__ W,
    const float* __restrict__ bias, const float* __restrict__ res,
    float* __restrict__ C, int M, int N, int Kpad, int act) {
  int wib = __builtin_amdgcn_readfirstlane(threadIdx.x >> 5);
  int wave = blockIdx.x * 4 + wib;
  int lane = threadIdx.x & 31;
  int mt_n = M >> 4;
  int ntp = (N + 15) >> 4;       // tiles over Npad
  int np_n = (ntp + 1) >> 1;     // tile pairs
  if (wave >= mt_n * np_n) return;
  int mt = wave / np_n, np = wave % np_n;
  int nt0 = np * 2, nt1 = nt0 + 1;
  bool has1 = (nt1 < ntp);
  int lan = lane & 15;
  int ha = (lane >= 16) ? 16 : 0;   // A 16x32 lane-group K-half
  int hb = (lane >= 16) ? 16 : 0;   // B 32x16 lane-group K-half
  const bf16_t* ap = A + (size_t)(mt * 16 + lan) * Kpad + ha;
  const bf16_t* bp0 = W + (size_t)(nt0 * 16 + lan) * Kpad + hb;
  v8f acc0 = {0.f, 0.f, 0.f, 0.f, 0.f, 0.f, 0.f, 0.f};
  v8f acc1 = acc0;
  if (has1) {
    const bf16_t* bp1 = W + (size_t)(nt1 * 16 + lan) * Kpad + hb;
    for (int k0 = 0; k0 < Kpad; k0 += 32) {
      v16bf av = cat8(*(const v8bf*)(ap + k0), *(const v8bf*)(ap + k0 + 8));
      v16bf bv0 = cat8(*(const v8bf*)(bp0 + k0), *(const v8bf*)(bp0 + k0 + 8));
      v16bf bv1 = cat8(*(const v8bf*)(bp1 + k0), *(const v8bf*)(bp1 + k0 + 8));
      acc0 = __builtin_amdgcn_wmma_f32_16x16x32_bf16(false, av, false, bv0, (short)0, acc0,
                                                     false, false);
      acc1 = __builtin_amdgcn_wmma_f32_16x16x32_bf16(false, av, false, bv1, (short)0, acc1,
                                                     false, false);
    }
  } else {
    for (int k0 = 0; k0 < Kpad; k0 += 32) {
      v16bf av = cat8(*(const v8bf*)(ap + k0), *(const v8bf*)(ap + k0 + 8));
      v16bf bv0 = cat8(*(const v8bf*)(bp0 + k0), *(const v8bf*)(bp0 + k0 + 8));
      acc0 = __builtin_amdgcn_wmma_f32_16x16x32_bf16(false, av, false, bv0, (short)0, acc0,
                                                     false, false);
    }
  }
  int row0 = mt * 16 + ((lane >= 16) ? 8 : 0);
  {
    int col = nt0 * 16 + lan;
    if (col < N) {
#pragma unroll
      for (int r = 0; r < 8; ++r) {
        int row = row0 + r;
        float v = acc0[r];
        if (bias) v += bias[col];
        if (res) v += res[(size_t)row * N + col];
        if (act) v = fmaxf(v, 0.f);
        C[(size_t)row * N + col] = v;
      }
    }
  }
  if (has1) {
    int col = nt1 * 16 + lan;
    if (col < N) {
#pragma unroll
      for (int r = 0; r < 8; ++r) {
        int row = row0 + r;
        float v = acc1[r];
        if (bias) v += bias[col];
        if (res) v += res[(size_t)row * N + col];
        if (act) v = fmaxf(v, 0.f);
        C[(size_t)row * N + col] = v;
      }
    }
  }
}

// ---------------------------------------------------------------------------
// Fused GRU step. h_in_bf [NB, OPAD] bf16 (pad cols zero), whh [912, OPAD] bf16,
// gi [NB*128, 900] f32 (bias included), gh bias bhh f32 [900].
// Writes h_out f32 [NB,300] and bf16 [NB,OPAD] (pad cols untouched == zero).
// ---------------------------------------------------------------------------
__global__ __launch_bounds__(128) void gru_step_kernel(
    const float* __restrict__ h_in_f, const bf16_t* __restrict__ h_in_bf,
    const float* __restrict__ gi, int t,
    const bf16_t* __restrict__ whh, const float* __restrict__ bhh,
    float* __restrict__ h_out_f, bf16_t* __restrict__ h_out_bf,
    float* __restrict__ e_acc, int e_off) {
  const int H = HID;
  int wib = __builtin_amdgcn_readfirstlane(threadIdx.x >> 5);
  int wave = blockIdx.x * 4 + wib;
  int lane = threadIdx.x & 31;
  const int nt_n = (H + 15) >> 4;  // 19
  if (wave >= (NB / 16) * nt_n) return;
  int mt = wave / nt_n, nt = wave % nt_n;
  int lan = lane & 15;
  int ha = (lane >= 16) ? 16 : 0;
  int hb = (lane >= 16) ? 16 : 0;
  int bcol = nt * 16 + lan;
  const bf16_t* ap = h_in_bf + (size_t)(mt * 16 + lan) * OPAD + ha;
  const bf16_t* bpr = whh + (size_t)(bcol)         * OPAD + hb;
  const bf16_t* bpz = whh + (size_t)(bcol + H)     * OPAD + hb;
  const bf16_t* bpn = whh + (size_t)(bcol + 2 * H) * OPAD + hb;
  v8f ar = {0.f, 0.f, 0.f, 0.f, 0.f, 0.f, 0.f, 0.f};
  v8f az = ar, an = ar;
  for (int k0 = 0; k0 < OPAD; k0 += 32) {
    v16bf av = cat8(*(const v8bf*)(ap + k0), *(const v8bf*)(ap + k0 + 8));
    v16bf br = cat8(*(const v8bf*)(bpr + k0), *(const v8bf*)(bpr + k0 + 8));
    v16bf bz = cat8(*(const v8bf*)(bpz + k0), *(const v8bf*)(bpz + k0 + 8));
    v16bf bn = cat8(*(const v8bf*)(bpn + k0), *(const v8bf*)(bpn + k0 + 8));
    ar = __builtin_amdgcn_wmma_f32_16x16x32_bf16(false, av, false, br, (short)0, ar, false, false);
    az = __builtin_amdgcn_wmma_f32_16x16x32_bf16(false, av, false, bz, (short)0, az, false, false);
    an = __builtin_amdgcn_wmma_f32_16x16x32_bf16(false, av, false, bn, (short)0, an, false, false);
  }
  int row0 = mt * 16 + ((lane >= 16) ? 8 : 0);
  int col = nt * 16 + lan;
  if (col < H) {
#pragma unroll
    for (int r = 0; r < 8; ++r) {
      int b = row0 + r;
      size_t girow = ((size_t)b * NA_G + t) * (3 * H);
      float gr = gi[girow + col];
      float gz = gi[girow + H + col];
      float gn = gi[girow + 2 * H + col];
      float rg = dsigmoid(gr + ar[r] + bhh[col]);
      float zg = dsigmoid(gz + az[r] + bhh[H + col]);
      float ng = tanhf(gn + rg * (an[r] + bhh[2 * H + col]));
      float hp = h_in_f[(size_t)b * H + col];
      float hn = (1.f - zg) * ng + zg * hp;
      h_out_f[(size_t)b * H + col] = hn;
      h_out_bf[(size_t)b * OPAD + col] = f2bf(hn);
      if (e_acc) e_acc[(size_t)b * 600 + e_off + col] += hn;
    }
  }
}

// ---------------------------------------------------------------------------
// Message attention: per (node, head), deg<=8 keys gathered via idx.
// Output written as bf16 [Nq, OPAD] (GEMM operand).
// ---------------------------------------------------------------------------
__global__ void msg_attn_kernel(const float* __restrict__ Q, const float* __restrict__ Kb,
                                const float* __restrict__ Vb, const int* __restrict__ idx,
                                bf16_t* __restrict__ O, int Nq, int deg, int heads) {
  int tid = blockIdx.x * blockDim.x + threadIdx.x;
  if (tid >= Nq * heads) return;
  int n = tid / heads, h = tid % heads;
  int dk = HID / heads;
  const float* q = Q + (size_t)n * HID + h * dk;
  float scale = rsqrtf((float)dk);
  int ei[8];
  float s[8];
  float mx = -1e30f;
  for (int j = 0; j < deg; ++j) {
    ei[j] = idx[(size_t)n * deg + j];
    const float* kp = Kb + (size_t)ei[j] * HID + h * dk;
    float a = 0.f;
    for (int d = 0; d < dk; ++d) a += q[d] * kp[d];
    s[j] = a * scale;
    mx = fmaxf(mx, s[j]);
  }
  float den = 0.f;
  for (int j = 0; j < deg; ++j) { s[j] = expf(s[j] - mx); den += s[j]; }
  float inv = 1.f / den;
  bf16_t* o = O + (size_t)n * OPAD + h * dk;
  for (int d = 0; d < dk; ++d) {
    float a = 0.f;
    for (int j = 0; j < deg; ++j) a += s[j] * Vb[(size_t)ei[j] * HID + h * dk + d];
    o[d] = f2bf(a * inv);
  }
}

// ---------------------------------------------------------------------------
// Readout attention: [B,128] queries x [B,16] keys, 6 heads, dk=50, masked.
// ---------------------------------------------------------------------------
__global__ void ro_attn_kernel(const float* __restrict__ Q, const float* __restrict__ Kb,
                               const float* __restrict__ Vb, const unsigned char* __restrict__ msk,
                               bf16_t* __restrict__ O) {
  int tid = blockIdx.x * blockDim.x + threadIdx.x;
  if (tid >= NB * NA_G * 6) return;
  int h = tid % 6;
  int i = (tid / 6) % NA_G;
  int b = tid / (6 * NA_G);
  const int dk = 50;
  const float* q = Q + ((size_t)b * NA_G + i) * HID + h * dk;
  float scale = rsqrtf(50.f);
  float s[NP_G];
  float mx = -1e30f;
  for (int j = 0; j < NP_G; ++j) {
    const float* kp = Kb + ((size_t)b * NP_G + j) * HID + h * dk;
    float a = 0.f;
    for (int d = 0; d < dk; ++d) a += q[d] * kp[d];
    a *= scale;
    if (msk[((size_t)b * NA_G + i) * NP_G + j]) a = -1e9f;
    s[j] = a;
    mx = fmaxf(mx, a);
  }
  float den = 0.f;
  for (int j = 0; j < NP_G; ++j) { s[j] = expf(s[j] - mx); den += s[j]; }
  float inv = 1.f / den;
  bf16_t* o = O + ((size_t)b * NA_G + i) * OPAD + h * dk;
  for (int d = 0; d < dk; ++d) {
    float a = 0.f;
    for (int j = 0; j < NP_G; ++j) a += s[j] * Vb[((size_t)b * NP_G + j) * HID + h * dk + d];
    o[d] = f2bf(a * inv);
  }
}

__global__ void mask_kernel(const float* __restrict__ a, const float* __restrict__ p,
                            unsigned char* __restrict__ m) {
  int tid = blockIdx.x * blockDim.x + threadIdx.x;
  if (tid >= NB * NA_G * NP_G) return;
  int j = tid & (NP_G - 1);
  int i = (tid >> 4) & (NA_G - 1);
  int b = tid >> 11;
  const float* ar = a + ((size_t)b * NA_G + i) * HID;
  const float* pr = p + ((size_t)b * NP_G + j) * HID;
  int any = 0;
  for (int h = 0; h < HID && !any; ++h) any = (ar[h] != 0.f) && (pr[h] != 0.f);
  m[tid] = any ? 0 : 1;  // 1 == masked
}

// ---------------------------------------------------------------------------
// Helpers (bf16 destinations are padded GEMM operands)
// ---------------------------------------------------------------------------
__global__ void edge_msg_kernel(const float* __restrict__ fh, const float* __restrict__ he,
                                const int* __restrict__ src, bf16_t* __restrict__ out, int E) {
  int tid = blockIdx.x * blockDim.x + threadIdx.x;
  if (tid >= E * OPAD) return;
  int e = tid / OPAD, h = tid % OPAD;
  float v = 0.f;
  if (h < HID)
    v = fh[(size_t)src[e] * HID + h] - he[(size_t)(e ^ 1) * HID + h];
  out[tid] = f2bf(v);
}

__global__ void sum_mail_bf(const float* __restrict__ he, const int* __restrict__ mb,
                            int deg, bf16_t* __restrict__ dst, int N, int ld, int off) {
  int tid = blockIdx.x * blockDim.x + threadIdx.x;
  if (tid >= N * HID) return;
  int n = tid / HID, h = tid % HID;
  float a = 0.f;
  for (int j = 0; j < deg; ++j) a += he[(size_t)mb[(size_t)n * deg + j] * HID + h];
  dst[(size_t)n * ld + off + h] = f2bf(a);
}

__global__ void copy_block_bf(const float* __restrict__ src, bf16_t* __restrict__ dst,
                              int N, int ld, int off) {
  int tid = blockIdx.x * blockDim.x + threadIdx.x;
  if (tid >= N * HID) return;
  int n = tid / HID, h = tid % HID;
  dst[(size_t)n * ld + off + h] = f2bf(src[(size_t)n * HID + h]);
}

__global__ void gather_block_bf(const float* __restrict__ src, const int* __restrict__ idx,
                                bf16_t* __restrict__ dst, int N, int ld, int off) {
  int tid = blockIdx.x * blockDim.x + threadIdx.x;
  if (tid >= N * HID) return;
  int n = tid / HID, h = tid % HID;
  dst[(size_t)n * ld + off + h] = f2bf(src[(size_t)idx[n] * HID + h]);
}

__global__ void rowmax_kernel(const float* __restrict__ x, float* __restrict__ h0) {
  int tid = blockIdx.x * blockDim.x + threadIdx.x;
  if (tid >= NB * HID) return;
  int b = tid / HID, c = tid % HID;
  float m = -1e30f;
  for (int i = 0; i < NA_G; ++i) m = fmaxf(m, x[((size_t)b * NA_G + i) * HID + c]);
  h0[tid] = m;
}

__global__ void zero_kernel(float* __restrict__ p, int n) {
  int i = blockIdx.x * blockDim.x + threadIdx.x;
  if (i < n) p[i] = 0.f;
}

__global__ void emb_bf_kernel(const float* __restrict__ e1, const float* __restrict__ e2,
                              bf16_t* __restrict__ emb) {  // [NB, 1216] padded
  int tid = blockIdx.x * blockDim.x + threadIdx.x;
  if (tid >= NB * 1216) return;
  int b = tid / 1216, c = tid % 1216;
  float v = 0.f;
  if (c < 600) v = e1[(size_t)b * 600 + c] * (1.f / 16.f);
  else if (c < 1200) v = e2[(size_t)b * 600 + (c - 600)] * (1.f / 16.f);
  emb[tid] = f2bf(v);
}

__global__ void out2_kernel(const float* __restrict__ x, const float* __restrict__ w,
                            const float* __restrict__ b, float* __restrict__ out) {
  int i = blockIdx.x * blockDim.x + threadIdx.x;
  if (i >= NB) return;
  float a = b[0];
  for (int k = 0; k < HID; ++k) a += x[(size_t)i * HID + k] * w[k];
  out[i] = a;
}

// ---------------------------------------------------------------------------
// Host orchestration
// ---------------------------------------------------------------------------
static inline int kpad32(int K) { return (K + 31) & ~31; }
static inline int npad16(int N) { return (N + 15) & ~15; }

static void gemm(hipStream_t st, const bf16_t* A, const bf16_t* W, const float* bias,
                 const float* res, float* C, int M, int N, int K, int act) {
  int ntp = (N + 15) / 16;
  int tiles = (M / 16) * ((ntp + 1) / 2);
  int blocks = (tiles + 3) / 4;
  gemm_kernel<<<blocks, 128, 0, st>>>(A, W, bias, res, C, M, N, kpad32(K), act);
}

extern "C" void kernel_launch(void* const* d_in, const int* in_sizes, int n_in,
                              void* d_out, int out_size, void* d_ws, size_t ws_size,
                              hipStream_t stream) {
  (void)in_sizes; (void)out_size; (void)ws_size;
  const size_t NAHf = (size_t)NAT * HID;
  const size_t EAHf = (size_t)EA * HID;
  const size_t NPHf = (size_t)NPH_N * HID;
  const size_t EPHf = (size_t)EP * HID;
  const size_t BHf = (size_t)NB * HID;

  const float* feat_a = (const float*)d_in[0];
  const float* feat_b = (const float*)d_in[1];
  const float* feat_p = (const float*)d_in[2];
  const float* feat_r = (const float*)d_in[3];
  const float* feat_ja = (const float*)d_in[4];
  const float* feat_jp = (const float*)d_in[5];
  const int* src_a = (const int*)d_in[6];
  const int* mb_a = (const int*)d_in[7];
  const int* src_p = (const int*)d_in[8];
  const int* mb_p = (const int*)d_in[9];
  const int* mb_ajp = (const int*)d_in[10];
  const int* pja_src = (const int*)d_in[11];
  const float* P[128];
  for (int i = 12; i < n_in && i < 128; ++i) P[i] = (const float*)d_in[i];

  char* cur = (char*)d_ws;
  auto alloc_bytes = [&](size_t b) -> void* {
    void* p = (void*)cur;
    cur += (b + 255) & ~(size_t)255;
    return p;
  };
  auto ew = [&](long long n) { return dim3((unsigned)((n + 255) / 256)); };

  // --- weights -> zero-padded bf16 [Npad, Kpad] ---
  bf16_t* wb[128] = {};
  auto cvtw = [&](int idx, int N, int K) {
    int Np = npad16(N), Kp = kpad32(K);
    int total = Np * Kp;
    bf16_t* p = (bf16_t*)alloc_bytes((size_t)total * sizeof(bf16_t));
    cvt_pad_kernel<<<ew(total), 256, 0, stream>>>(P[idx], p, N, K, Kp, total);
    wb[idx] = p;
  };
  cvtw(12, HID, 42); cvtw(14, HID, 14); cvtw(16, HID, 194);
  cvtw(18, HID, 34); cvtw(20, HID, 236);
  const int attn_bases[4] = {22, 30, 38, 46};
  for (int a = 0; a < 4; ++a)
    for (int j = 0; j < 8; j += 2) cvtw(attn_bases[a] + j, HID, HID);
  cvtw(54, HID, HID); cvtw(56, HID, HID); cvtw(58, HID, HID); cvtw(60, HID, HID);
  cvtw(62, HID, 3 * HID); cvtw(64, HID, 3 * HID); cvtw(66, HID, 3 * HID);
  const int ro_bases[2] = {68, 84};
  for (int a = 0; a < 2; ++a) {
    int base = ro_bases[a];
    for (int j = 0; j < 8; j += 2) cvtw(base + j, HID, HID);
    cvtw(base + 8, 3 * HID, HID);   // fw wih
    cvtw(base + 9, 3 * HID, HID);   // fw whh
    cvtw(base + 12, 3 * HID, HID);  // bw wih
    cvtw(base + 13, 3 * HID, HID);  // bw whh
  }
  cvtw(100, HID, 4 * HID);
  cvtw(102, HID, HID);

  // --- buffers ---
  float* f_a  = (float*)alloc_bytes(NAHf * 4);
  float* f_p  = (float*)alloc_bytes(NPHf * 4);
  float* x_ab = (float*)alloc_bytes(EAHf * 4);
  float* x_pr = (float*)alloc_bytes(EPHf * 4);
  float* fh_a = (float*)alloc_bytes(NAHf * 4);
  float* fh_p = (float*)alloc_bytes(NPHf * 4);
  float* fjh_a = (float*)alloc_bytes(NAHf * 4);
  float* fjh_p = (float*)alloc_bytes(NPHf * 4);
  float* h_ab = (float*)alloc_bytes(EAHf * 4);
  float* h_pr = (float*)alloc_bytes(EPHf * 4);
  float* t0 = (float*)alloc_bytes(EAHf * 4);   // f32 scratch (k-proj / gi_fw)
  float* t1 = (float*)alloc_bytes(EAHf * 4);   // f32 scratch (v-proj / gi_bw)
  float* t2 = (float*)alloc_bytes(NAHf * 4);   // f32 scratch (q-proj / h)
  bf16_t* sA = (bf16_t*)alloc_bytes((size_t)EA * OPAD * sizeof(bf16_t));  // staged GEMM A
  float* hsAf = (float*)alloc_bytes(BHf * 4);
  float* hsBf = (float*)alloc_bytes(BHf * 4);
  bf16_t* hsAb = (bf16_t*)alloc_bytes((size_t)NB * OPAD * sizeof(bf16_t));
  bf16_t* hsBb = (bf16_t*)alloc_bytes((size_t)NB * OPAD * sizeof(bf16_t));
  float* h0b = (float*)alloc_bytes(BHf * 4);
  float* e1 = (float*)alloc_bytes((size_t)NB * 600 * 4);
  float* e2 = (float*)alloc_bytes((size_t)NB * 600 * 4);
  float* o0 = (float*)alloc_bytes(BHf * 4);
  float* o1 = (float*)alloc_bytes(BHf * 4);
  unsigned char* maskb = (unsigned char*)alloc_bytes((size_t)NB * NA_G * NP_G);

  // stage f32 [M,K] activations as bf16 [M,Kpad]
  auto stage = [&](bf16_t* dst, const float* src, int M, int K) {
    int Kp = kpad32(K);
    long long total = (long long)M * Kp;
    cvt_pad_kernel<<<ew(total), 256, 0, stream>>>(src, dst, M, K, Kp, (int)total);
  };

  // ---------------- init features ----------------
  stage(sA, feat_a, NAT, 42);
  gemm(stream, sA, wb[12], P[13], nullptr, f_a, NAT, HID, 42, 1);
  stage(sA, feat_b, EA, 14);
  gemm(stream, sA, wb[14], P[15], nullptr, x_ab, EA, HID, 14, 1);
  stage(sA, feat_p, NPH_N, 194);
  gemm(stream, sA, wb[16], P[17], nullptr, f_p, NPH_N, HID, 194, 1);
  stage(sA, feat_r, EP, 34);
  gemm(stream, sA, wb[18], P[19], nullptr, x_pr, EP, HID, 34, 1);
  stage(sA, feat_ja, NAT, 236);
  gemm(stream, sA, wb[20], P[21], nullptr, fjh_a, NAT, HID, 236, 1);
  stage(sA, feat_jp, NPH_N, 236);
  gemm(stream, sA, wb[20], P[21], nullptr, fjh_p, NPH_N, HID, 236, 1);
  hipMemcpyAsync(fh_a, f_a, NAHf * 4, hipMemcpyDeviceToDevice, stream);
  hipMemcpyAsync(fh_p, f_p, NPHf * 4, hipMemcpyDeviceToDevice, stream);
  hipMemcpyAsync(h_ab, x_ab, EAHf * 4, hipMemcpyDeviceToDevice, stream);
  hipMemcpyAsync(h_pr, x_pr, EPHf * 4, hipMemcpyDeviceToDevice, stream);

  // one attention block: upd <- lin_o(attn(lin_q(upd); lin_k/v(src)[idx])) + upd
  auto attn = [&](int base, float* upd, int Nq, const float* srcT, int Ns,
                  const int* idx, int deg) {
    stage(sA, upd, Nq, HID);
    gemm(stream, sA, wb[base + 0], P[base + 1], nullptr, t2, Nq, HID, HID, 0);   // q
    stage(sA, srcT, Ns, HID);
    gemm(stream, sA, wb[base + 2], P[base + 3], nullptr, t0, Ns, HID, HID, 0);   // k
    gemm(stream, sA, wb[base + 4], P[base + 5], nullptr, t1, Ns, HID, HID, 0);   // v
    pad_zero_bf<<<ew((long long)Nq * (OPAD - HID)), 256, 0, stream>>>(sA, Nq, OPAD, HID);
    msg_attn_kernel<<<ew((long long)Nq * 4), 256, 0, stream>>>(t2, t0, t1, idx, sA, Nq, deg, 4);
    gemm(stream, sA, wb[base + 6], P[base + 7], upd, upd, Nq, HID, HID, 0);      // o + residual
  };

  // ---------------- MVMP loop (DEPTH-1 = 2) ----------------
  for (int it = 0; it < 2; ++it) {
    attn(22, fh_a, NAT, h_ab, EA, mb_a, 4);
    attn(30, fh_p, NPH_N, h_pr, EP, mb_p, 2);
    attn(38, fjh_p, NPH_N, fjh_a, NAT, mb_ajp, 8);
    attn(46, fjh_a, NAT, fjh_p, NPH_N, pja_src, 1);

    edge_msg_kernel<<<ew((long long)EA * OPAD), 256, 0, stream>>>(fh_a, h_ab, src_a, sA, EA);
    gemm(stream, sA, wb[54 + 2 * it], P[55 + 2 * it], x_ab, h_ab, EA, HID, HID, 1);
    edge_msg_kernel<<<ew((long long)EP * OPAD), 256, 0, stream>>>(fh_p, h_pr, src_p, sA, EP);
    gemm(stream, sA, wb[58 + 2 * it], P[59 + 2 * it], x_pr, h_pr, EP, HID, HID, 1);
  }

  // ---------------- final node updates (K=900, Kpad=928 concat in sA) ----------------
  const int K3 = 3 * HID, K3p = kpad32(3 * HID);
  sum_mail_bf<<<ew((long long)NAT * HID), 256, 0, stream>>>(h_ab, mb_a, 4, sA, NAT, K3p, 0);
  copy_block_bf<<<ew((long long)NAT * HID), 256, 0, stream>>>(fh_a, sA, NAT, K3p, 300);
  copy_block_bf<<<ew((long long)NAT * HID), 256, 0, stream>>>(f_a, sA, NAT, K3p, 600);
  pad_zero_bf<<<ew((long long)NAT * (K3p - K3)), 256, 0, stream>>>(sA, NAT, K3p, K3);
  gemm(stream, sA, wb[62], P[63], nullptr, fh_a, NAT, HID, K3, 0);

  sum_mail_bf<<<ew((long long)NPH_N * HID), 256, 0, stream>>>(h_pr, mb_p, 2, sA, NPH_N, K3p, 0);
  copy_block_bf<<<ew((long long)NPH_N * HID), 256, 0, stream>>>(fh_p, sA, NPH_N, K3p, 300);
  copy_block_bf<<<ew((long long)NPH_N * HID), 256, 0, stream>>>(f_p, sA, NPH_N, K3p, 600);
  pad_zero_bf<<<ew((long long)NPH_N * (K3p - K3)), 256, 0, stream>>>(sA, NPH_N, K3p, K3);
  gemm(stream, sA, wb[64], P[65], nullptr, fh_p, NPH_N, HID, K3, 0);

  sum_mail_bf<<<ew((long long)NPH_N * HID), 256, 0, stream>>>(fjh_a, mb_ajp, 8, sA, NPH_N, K3p, 0);
  copy_block_bf<<<ew((long long)NPH_N * HID), 256, 0, stream>>>(fjh_p, sA, NPH_N, K3p, 300);
  copy_block_bf<<<ew((long long)NPH_N * HID), 256, 0, stream>>>(f_p, sA, NPH_N, K3p, 600);
  pad_zero_bf<<<ew((long long)NPH_N * (K3p - K3)), 256, 0, stream>>>(sA, NPH_N, K3p, K3);
  gemm(stream, sA, wb[66], P[67], nullptr, fjh_p, NPH_N, HID, K3, 0);  // new fjh_p

  gather_block_bf<<<ew((long long)NAT * HID), 256, 0, stream>>>(fjh_p, pja_src, sA, NAT, K3p, 0);
  copy_block_bf<<<ew((long long)NAT * HID), 256, 0, stream>>>(fjh_a, sA, NAT, K3p, 300);
  copy_block_bf<<<ew((long long)NAT * HID), 256, 0, stream>>>(f_a, sA, NAT, K3p, 600);
  pad_zero_bf<<<ew((long long)NAT * (K3p - K3)), 256, 0, stream>>>(sA, NAT, K3p, K3);
  gemm(stream, sA, wb[66], P[67], nullptr, fjh_a, NAT, HID, K3, 0);

  // ---------------- readouts ----------------
  auto readout = [&](int base, const float* xa, const float* xp, float* e_out) {
    float* kro = t0;
    float* vro = t1;
    stage(sA, xa, NAT, HID);
    gemm(stream, sA, wb[base + 0], P[base + 1], nullptr, t2, NAT, HID, HID, 0);  // q -> t2
    stage(sA, xp, NPH_N, HID);
    gemm(stream, sA, wb[base + 2], P[base + 3], nullptr, kro, NPH_N, HID, HID, 0);
    gemm(stream, sA, wb[base + 4], P[base + 5], nullptr, vro, NPH_N, HID, HID, 0);
    mask_kernel<<<ew((long long)NB * NA_G * NP_G), 256, 0, stream>>>(xa, xp, maskb);
    pad_zero_bf<<<ew((long long)NAT * (OPAD - HID)), 256, 0, stream>>>(sA, NAT, OPAD, HID);
    ro_attn_kernel<<<ew((long long)NB * NA_G * 6), 256, 0, stream>>>(t2, kro, vro, maskb, sA);
    gemm(stream, sA, wb[base + 6], P[base + 7], xa, t2, NAT, HID, HID, 0);  // h = attn + a
    rowmax_kernel<<<ew((long long)NB * HID), 256, 0, stream>>>(t2, h0b);
    stage(sA, t2, NAT, HID);
    gemm(stream, sA, wb[base + 8], P[base + 10], nullptr, t0, NAT, 3 * HID, HID, 0);   // gi fw
    gemm(stream, sA, wb[base + 12], P[base + 14], nullptr, t1, NAT, 3 * HID, HID, 0);  // gi bw
    zero_kernel<<<ew(NB * 600), 256, 0, stream>>>(e_out, NB * 600);
    pad_zero_bf<<<ew((long long)NB * (OPAD - HID)), 256, 0, stream>>>(hsBb, NB, OPAD, HID);
    const int gru_blocks = ((NB / 16) * 19 + 3) / 4;
    // forward scan: only first NP_G outputs are consumed by the readout slice
    hipMemcpyAsync(hsAf, h0b, BHf * 4, hipMemcpyDeviceToDevice, stream);
    stage(hsAb, h0b, NB, HID);
    float* af = hsAf; float* bf = hsBf;
    bf16_t* ab = hsAb; bf16_t* bb = hsBb;
    for (int t = 0; t < NP_G; ++t) {
      gru_step_kernel<<<gru_blocks, 128, 0, stream>>>(af, ab, t0, t, wb[base + 9], P[base + 11],
                                                      bf, bb, e_out, 0);
      float* tf = af; af = bf; bf = tf;
      bf16_t* tb = ab; ab = bb; bb = tb;
    }
    // backward scan over reversed sequence (all 128 steps; last 16 accumulate)
    hipMemcpyAsync(hsAf, h0b, BHf * 4, hipMemcpyDeviceToDevice, stream);
    stage(hsAb, h0b, NB, HID);
    af = hsAf; bf = hsBf; ab = hsAb; bb = hsBb;
    for (int tp = 0; tp < NA_G; ++tp) {
      int t = NA_G - 1 - tp;
      float* acc = (tp >= NA_G - NP_G) ? e_out : nullptr;
      gru_step_kernel<<<gru_blocks, 128, 0, stream>>>(af, ab, t1, t, wb[base + 13], P[base + 15],
                                                      bf, bb, acc, 300);
      float* tf = af; af = bf; bf = tf;
      bf16_t* tb = ab; ab = bb; bb = tb;
    }
  };
  readout(68, fh_a, fh_p, e1);
  readout(84, fjh_a, fjh_p, e2);

  // ---------------- output head ----------------
  emb_bf_kernel<<<ew((long long)NB * 1216), 256, 0, stream>>>(e1, e2, sA);
  gemm(stream, sA, wb[100], P[101], nullptr, o0, NB, HID, 4 * HID, 1);
  stage(sA, o0, NB, HID);
  gemm(stream, sA, wb[102], P[103], nullptr, o1, NB, HID, HID, 1);
  out2_kernel<<<1, 256, 0, stream>>>(o1, P[104], P[105], (float*)d_out);
}
